// Seq2SeqLSTM_54150947668722
// MI455X (gfx1250) — matrix operand order repaired and checked by reading, not compiled, not run on hardware
//
#include <hip/hip_runtime.h>
#include <hip/hip_bf16.h>

// ---------------------------------------------------------------------------
// Seq2Seq LSTM for MI455X (gfx1250): f16 WMMA (16x16x32, f32 accum).
// v2: each wave owns one 16x16 (batch x hidden) tile for ALL FOUR gates ->
//     A-fragment loaded once per K-step (4x less activation traffic) and the
//     LSTM nonlinearity is applied directly from accumulator registers
//     (no LDS gate exchange, no block barrier).
// ---------------------------------------------------------------------------

typedef _Float16 half_t;
typedef __attribute__((ext_vector_type(16))) _Float16 v16h;
typedef __attribute__((ext_vector_type(8)))  _Float16 v8h;
typedef __attribute__((ext_vector_type(8)))  float    v8f;

#define B_  128
#define S_  256
#define I_  512
#define H_  1024
#define O_  512
#define T_  128

// ---------------------------------------------------------------------------
// WMMA helpers
// ---------------------------------------------------------------------------
__device__ inline v8f wmma_f16(v16h a, v16h b, v8f c) {
  return __builtin_amdgcn_wmma_f32_16x16x32_f16(
      /*neg_a=*/false, a, /*neg_b=*/false, b,
      /*c_mod=*/(short)0, c, /*reuse_a=*/false, /*reuse_b=*/false);
}

// A fragment: 16x32 f16 tile of a row-major matrix (rows = M, stride = rs).
// Lane L: m = L&15, kh = L>>4.  v[0..15] = K(kh*16 .. kh*16+15).
__device__ inline v16h load_a_frag(const half_t* __restrict__ base,
                                   int rs, int m0, int k0) {
  const int lane = threadIdx.x & 31;
  const int m    = lane & 15;
  const int kh   = lane >> 4;
  const half_t* p = base + (size_t)(m0 + m) * rs + k0 + kh * 16;
  v8h lo = *(const v8h*)(p);
  v8h hi = *(const v8h*)(p + 8);
  v16h r;
#pragma unroll
  for (int j = 0; j < 8; ++j) { r[j] = lo[j]; r[j + 8] = hi[j]; }
  return r;
}

// B fragment: 32x16 f16 tile where B[k][n] = W[n0+n][k0+k] (W row-major, ld=ldw).
// Lane L: n = L&15, kb = (L>>4)*16; 16 contiguous f16 per lane (32B load).
__device__ inline v16h load_b_frag(const half_t* __restrict__ W,
                                   int ldw, int n0, int k0) {
  const int lane = threadIdx.x & 31;
  const int n    = lane & 15;
  const int kb   = (lane >> 4) * 16;
  return *(const v16h*)(W + (size_t)(n0 + n) * ldw + k0 + kb);
}

__device__ inline float sigmoidf_(float x) { return 1.0f / (1.0f + __expf(-x)); }

// ---------------------------------------------------------------------------
// One LSTM step:  gates = inp @ Wih^T + h_prev @ Whh^T + bias ; cell update.
// Grid: (H/16, B/64).  Block: 128 threads = 4 waves; wave w owns m-tile
// m0 = blockIdx.y*64 + w*16 and ALL 4 gates (4 accumulators).
// inp:    f16, rows stride inp_rs, width K_in
// Wih:    f16 [4H][K_in] row-major (gate g rows at g*H)
// Whh:    f16 [4H][H]    row-major
// bias:   f32 [4H]  (b_ih + b_hh)
// h_prev: f16 [B][H];  c: f32 [B][H] (updated);  h_next: f16 [B][H]
// ---------------------------------------------------------------------------
__global__ __launch_bounds__(128) void lstm_step_kernel(
    const half_t* __restrict__ inp, int inp_rs, int K_in,
    const half_t* __restrict__ Wih, const half_t* __restrict__ Whh,
    const float*  __restrict__ bias,
    const half_t* __restrict__ h_prev,
    float*        __restrict__ c,
    half_t*       __restrict__ h_next)
{
  const int n0   = blockIdx.x * 16;                              // hidden cols
  const int m0   = blockIdx.y * 64 + (threadIdx.x >> 5) * 16;    // batch rows
  const int lane = threadIdx.x & 31;

  v8f acc[4] = {{}, {}, {}, {}};   // gate order: i, f, g, o

  // inp @ Wih^T : one A load feeds 4 gate WMMAs
  for (int k0 = 0; k0 < K_in; k0 += 32) {
    v16h a = load_a_frag(inp, inp_rs, m0, k0);
#pragma unroll
    for (int g = 0; g < 4; ++g) {
      v16h b = load_b_frag(Wih + (size_t)g * H_ * K_in, K_in, n0, k0);
      acc[g] = wmma_f16(a, b, acc[g]);
    }
  }
  // h_prev @ Whh^T
  for (int k0 = 0; k0 < H_; k0 += 32) {
    v16h a = load_a_frag(h_prev, H_, m0, k0);
#pragma unroll
    for (int g = 0; g < 4; ++g) {
      v16h b = load_b_frag(Whh + (size_t)g * H_ * H_, H_, n0, k0);
      acc[g] = wmma_f16(a, b, acc[g]);
    }
  }

  // Epilogue straight from registers: lane holds i/f/g/o for its 8 elements.
  // D layout: col n = lane&15, rows m = (lane>>4)*8 + j.
  const int n   = lane & 15;
  const int mb  = (lane >> 4) * 8;
  const int col = n0 + n;
  const float bi = bias[0 * H_ + col];
  const float bf = bias[1 * H_ + col];
  const float bg = bias[2 * H_ + col];
  const float bo = bias[3 * H_ + col];
#pragma unroll
  for (int j = 0; j < 8; ++j) {
    const size_t idx = (size_t)(m0 + mb + j) * H_ + col;
    const float gi = acc[0][j] + bi;
    const float gf = acc[1][j] + bf;
    const float gg = acc[2][j] + bg;
    const float go = acc[3][j] + bo;
    const float cnew = sigmoidf_(gf) * c[idx] + sigmoidf_(gi) * tanhf(gg);
    c[idx]      = cnew;
    h_next[idx] = (half_t)(sigmoidf_(go) * tanhf(cnew));
  }
}

// ---------------------------------------------------------------------------
// Decoder projection: y = h @ Wproj^T + b_proj.
// Writes fp32 y into d_out[b][t][:] and f16 y into inp_next (next-step input).
// Grid: (O/16, B/16). Block: 32 threads (1 wave).
// ---------------------------------------------------------------------------
__global__ __launch_bounds__(32) void proj_kernel(
    const half_t* __restrict__ h,
    const half_t* __restrict__ Wproj,   // f16 [O][H]
    const float*  __restrict__ b_proj,  // f32 [O]
    float*        __restrict__ y_out,   // [B][T][O]
    half_t*       __restrict__ inp_next,// f16 [B][O]
    int t)
{
  const int n0   = blockIdx.x * 16;
  const int m0   = blockIdx.y * 16;
  const int lane = threadIdx.x & 31;

  v8f acc = {};
  for (int k0 = 0; k0 < H_; k0 += 32) {
    v16h a = load_a_frag(h, H_, m0, k0);
    v16h b = load_b_frag(Wproj, H_, n0, k0);
    acc = wmma_f16(a, b, acc);
  }

  const int n  = lane & 15;
  const int mb = (lane >> 4) * 8;
  const float bv = b_proj[n0 + n];
#pragma unroll
  for (int j = 0; j < 8; ++j) {
    const int m = mb + j;
    const float v = acc[j] + bv;
    y_out[((size_t)(m0 + m) * T_ + t) * O_ + (n0 + n)] = v;
    inp_next[(size_t)(m0 + m) * O_ + (n0 + n)] = (half_t)v;
  }
}

// ---------------------------------------------------------------------------
// Setup kernels
// ---------------------------------------------------------------------------
__global__ void cvt_f32_to_f16_kernel(const float* __restrict__ s,
                                      half_t* __restrict__ d, int n) {
  int i = blockIdx.x * blockDim.x + threadIdx.x;
  const int stride = gridDim.x * blockDim.x;
  for (; i < n; i += stride) d[i] = (half_t)s[i];
}

__global__ void add_bias_kernel(const float* __restrict__ a,
                                const float* __restrict__ b,
                                float* __restrict__ out, int n) {
  int i = blockIdx.x * blockDim.x + threadIdx.x;
  if (i < n) out[i] = a[i] + b[i];
}

// ---------------------------------------------------------------------------
// Host orchestration
// ---------------------------------------------------------------------------
extern "C" void kernel_launch(void* const* d_in, const int* in_sizes, int n_in,
                              void* d_out, int out_size, void* d_ws, size_t ws_size,
                              hipStream_t stream) {
  (void)in_sizes; (void)n_in; (void)out_size; (void)ws_size;

  const float* x        = (const float*)d_in[0];
  // d_in[1] = tgt_len (scalar) -> fixed T_ = 128
  const float* W_ih_enc = (const float*)d_in[2];
  const float* W_hh_enc = (const float*)d_in[3];
  const float* b_ih_enc = (const float*)d_in[4];
  const float* b_hh_enc = (const float*)d_in[5];
  const float* W_ih_dec = (const float*)d_in[6];
  const float* W_hh_dec = (const float*)d_in[7];
  const float* b_ih_dec = (const float*)d_in[8];
  const float* b_hh_dec = (const float*)d_in[9];
  const float* W_proj   = (const float*)d_in[10];
  const float* b_proj   = (const float*)d_in[11];
  float* out = (float*)d_out;

  // ---- workspace layout (256B aligned) ----
  char*  w   = (char*)d_ws;
  size_t off = 0;
  auto alloc = [&](size_t bytes) -> void* {
    void* p = w + off;
    off = (off + bytes + 255) & ~(size_t)255;
    return p;
  };
  half_t* x_h       = (half_t*)alloc((size_t)B_ * S_ * I_ * 2);
  half_t* Wih_enc_h = (half_t*)alloc((size_t)4 * H_ * I_ * 2);
  half_t* Whh_enc_h = (half_t*)alloc((size_t)4 * H_ * H_ * 2);
  half_t* Wih_dec_h = (half_t*)alloc((size_t)4 * H_ * O_ * 2);
  half_t* Whh_dec_h = (half_t*)alloc((size_t)4 * H_ * H_ * 2);
  half_t* Wproj_h   = (half_t*)alloc((size_t)O_ * H_ * 2);
  float*  b_enc     = (float*) alloc((size_t)4 * H_ * 4);
  float*  b_dec     = (float*) alloc((size_t)4 * H_ * 4);
  half_t* h_a       = (half_t*)alloc((size_t)B_ * H_ * 2);
  half_t* h_b       = (half_t*)alloc((size_t)B_ * H_ * 2);
  float*  c_buf     = (float*) alloc((size_t)B_ * H_ * 4);
  half_t* inp_h     = (half_t*)alloc((size_t)B_ * O_ * 2);

  // ---- precision conversion + bias folding ----
  auto cvt = [&](const float* s, half_t* d, int n) {
    int blocks = (n + 255) / 256; if (blocks > 4096) blocks = 4096;
    cvt_f32_to_f16_kernel<<<blocks, 256, 0, stream>>>(s, d, n);
  };
  cvt(x,        x_h,       B_ * S_ * I_);
  cvt(W_ih_enc, Wih_enc_h, 4 * H_ * I_);
  cvt(W_hh_enc, Whh_enc_h, 4 * H_ * H_);
  cvt(W_ih_dec, Wih_dec_h, 4 * H_ * O_);
  cvt(W_hh_dec, Whh_dec_h, 4 * H_ * H_);
  cvt(W_proj,   Wproj_h,   O_ * H_);
  add_bias_kernel<<<(4 * H_ + 255) / 256, 256, 0, stream>>>(b_ih_enc, b_hh_enc, b_enc, 4 * H_);
  add_bias_kernel<<<(4 * H_ + 255) / 256, 256, 0, stream>>>(b_ih_dec, b_hh_dec, b_dec, 4 * H_);

  // ---- state init (every call: graph replays must be deterministic) ----
  hipMemsetAsync(h_a,   0, (size_t)B_ * H_ * 2, stream);
  hipMemsetAsync(c_buf, 0, (size_t)B_ * H_ * 4, stream);
  hipMemsetAsync(inp_h, 0, (size_t)B_ * O_ * 2, stream);

  const dim3 grid_step(H_ / 16, B_ / 64);   // 64 x 2, 4 waves/block
  const dim3 grid_proj(O_ / 16, B_ / 16);   // 32 x 8, 1 wave/block

  // ---- encoder: 256 sequential steps (stream ordering = recurrence dep) ----
  half_t* hp = h_a;
  half_t* hn = h_b;
  for (int t = 0; t < S_; ++t) {
    lstm_step_kernel<<<grid_step, 128, 0, stream>>>(
        x_h + (size_t)t * I_, S_ * I_, I_,
        Wih_enc_h, Whh_enc_h, b_enc, hp, c_buf, hn);
    half_t* tmp = hp; hp = hn; hn = tmp;
  }

  // ---- decoder: 128 autoregressive steps; (h,c) carried from encoder ----
  for (int t = 0; t < T_; ++t) {
    lstm_step_kernel<<<grid_step, 128, 0, stream>>>(
        inp_h, O_, O_,
        Wih_dec_h, Whh_dec_h, b_dec, hp, c_buf, hn);
    proj_kernel<<<grid_proj, 32, 0, stream>>>(
        hn, Wproj_h, b_proj, out, inp_h, t);
    half_t* tmp = hp; hp = hn; hn = tmp;
  }
}
